// Attention_82867099009253
// MI455X (gfx1250) — matrix operand order repaired and checked by reading, not compiled, not run on hardware
//
#include <hip/hip_runtime.h>

// ---------------------------------------------------------------------------
// Problem constants (match reference)
// ---------------------------------------------------------------------------
#define BB   4
#define LL   2000
#define LP   2016          // key dim padded to multiple of 32
#define CC   512
#define C2   1024
#define HH   8
#define HD   64
#define CHUNK 1000
#define SCALE 0.125f       // hd^-0.5 = 64^-0.5

typedef __attribute__((ext_vector_type(16))) _Float16 v16h;
typedef __attribute__((ext_vector_type(8)))  float    v8f;

union U16h { v16h v; uint4 q[2]; };

#define DEVI static __device__ __forceinline__

// ---------------------------------------------------------------------------
// Fragment loaders, conforming to CDNA5 ISA 7.12.2 layouts.
//   A (16x32 f16): lane<16: M=lane, K={0..7,16..23}; lane>=16: M=lane-16,
//                  K={8..15,24..31}  -> two 16B loads from row-major [M][K].
//   B (32x16 f16): lane<16: N=lane, K={0..7,16..23}; lane>=16: N=lane-16,
//                  K={8..15,24..31} -> two 16B loads from [N][K]-major storage.
// ---------------------------------------------------------------------------
DEVI v16h loadA(const _Float16* A, int ld, int mbase, int kbase) {
  int ln = threadIdx.x & 31;
  const _Float16* p = A + (size_t)(mbase + (ln & 15)) * ld + kbase + ((ln >> 4) << 3);
  U16h u;
  u.q[0] = *reinterpret_cast<const uint4*>(p);
  u.q[1] = *reinterpret_cast<const uint4*>(p + 16);
  return u.v;
}

DEVI v16h loadB(const _Float16* Bt, int ld, int nbase, int kbase) {
  int ln = threadIdx.x & 31;
  const _Float16* p = Bt + (size_t)(nbase + (ln & 15)) * ld + kbase + ((ln >> 4) << 3);
  U16h u;
  u.q[0] = *reinterpret_cast<const uint4*>(p);
  u.q[1] = *reinterpret_cast<const uint4*>(p + 16);
  return u.v;
}

DEVI v8f wmma16(v16h a, v16h b, v8f c) {
  return __builtin_amdgcn_wmma_f32_16x16x32_f16(false, a, false, b, (short)0, c,
                                                false, false);
}

// ---------------------------------------------------------------------------
// Kernel 1: depthwise chunked conv (k=3, pad=1 per CHUNK) + residual + bias,
// emit f16 kv_in [B][LP][C] (zero pad rows) and f16 x [B][LL][C].
// ---------------------------------------------------------------------------
__global__ __launch_bounds__(256) void prep_conv(const float* __restrict__ x,
                                                 const float* __restrict__ Wl,
                                                 const float* __restrict__ bl,
                                                 _Float16* __restrict__ xh,
                                                 _Float16* __restrict__ kvin) {
  int i = blockIdx.x * 256 + threadIdx.x;
  const int total = BB * LP * CC;
  if (i >= total) return;
  int c = i % CC;
  int l = (i / CC) % LP;
  int b = i / (CC * LP);
  if (l < LL) {
    size_t base = ((size_t)b * LL + l) * CC + c;
    float x0 = x[base];
    int lc = l % CHUNK;
    float xm = (lc > 0)         ? x[base - CC] : 0.0f;
    float xp = (lc < CHUNK - 1) ? x[base + CC] : 0.0f;
    float y = Wl[c * 3 + 0] * xm + Wl[c * 3 + 1] * x0 + Wl[c * 3 + 2] * xp
            + bl[c] + x0;
    kvin[((size_t)b * LP + l) * CC + c] = (_Float16)y;
    xh[base] = (_Float16)x0;
  } else {
    kvin[((size_t)b * LP + l) * CC + c] = (_Float16)0.0f;
  }
}

// ---------------------------------------------------------------------------
// Kernel 2: f32 -> f16 weight convert (Wq then Wkv, both [out][in]-major,
// which is exactly the B-operand-friendly layout).
// ---------------------------------------------------------------------------
__global__ __launch_bounds__(256) void prep_weights(const float* __restrict__ Wq,
                                                    const float* __restrict__ Wkv,
                                                    _Float16* __restrict__ Wqh,
                                                    _Float16* __restrict__ Wkvh) {
  int i = blockIdx.x * 256 + threadIdx.x;
  const int nq = CC * CC;
  const int nkv = C2 * CC;
  if (i < nq) Wqh[i] = (_Float16)Wq[i];
  else if (i < nq + nkv) Wkvh[i - nq] = (_Float16)Wkv[i - nq];
}

// ---------------------------------------------------------------------------
// Kernel 3: q = x @ Wq^T + bq   (M=8000, N=512, K=512), f16 out token-major.
// One wave -> 16x64 tile. 500 blocks x 8 waves.
// ---------------------------------------------------------------------------
__global__ __launch_bounds__(256) void gemm_q(const _Float16* __restrict__ xh,
                                              const _Float16* __restrict__ Wqh,
                                              const float* __restrict__ bq,
                                              _Float16* __restrict__ qh) {
  int wid = blockIdx.x * 8 + (threadIdx.x >> 5);
  int mtile = wid >> 3;            // 0..499
  int ntile = wid & 7;             // 0..7
  int mbase = mtile * 16;
  int nbase = ntile * 64;
  int ln = threadIdx.x & 31;
  v8f acc[4] = {};
  for (int kk = 0; kk < CC; kk += 32) {
    v16h a = loadA(xh, CC, mbase, kk);
#pragma unroll
    for (int j = 0; j < 4; ++j) {
      v16h b = loadB(Wqh, CC, nbase + 16 * j, kk);
      acc[j] = wmma16(a, b, acc[j]);
    }
  }
  int g = ln >> 4, nl = ln & 15;
#pragma unroll
  for (int j = 0; j < 4; ++j) {
    int n = nbase + 16 * j + nl;
    float bias = bq[n];
#pragma unroll
    for (int r = 0; r < 8; ++r) {
      int m = mbase + r + 8 * g;
      qh[(size_t)m * CC + n] = (_Float16)(acc[j][r] + bias);
    }
  }
}

// ---------------------------------------------------------------------------
// Kernel 4: kv = kv_in @ Wkv^T + bkv (M=8064, N=1024, K=512).
// First 512 cols -> k token-major [B][LP][C]; last 512 -> V TRANSPOSED
// [B][H][HD][LP] so attention's PV B-fragments are contiguous.
// 1008 blocks x 8 waves.
// ---------------------------------------------------------------------------
__global__ __launch_bounds__(256) void gemm_kv(const _Float16* __restrict__ kvin,
                                               const _Float16* __restrict__ Wkvh,
                                               const float* __restrict__ bkv,
                                               _Float16* __restrict__ kh,
                                               _Float16* __restrict__ vT) {
  int wid = blockIdx.x * 8 + (threadIdx.x >> 5);
  int mtile = wid >> 4;            // 0..503
  int ntile = wid & 15;            // 0..15
  int mbase = mtile * 16;
  int nbase = ntile * 64;
  int ln = threadIdx.x & 31;
  v8f acc[4] = {};
  for (int kk = 0; kk < CC; kk += 32) {
    v16h a = loadA(kvin, CC, mbase, kk);
#pragma unroll
    for (int j = 0; j < 4; ++j) {
      v16h b = loadB(Wkvh, CC, nbase + 16 * j, kk);
      acc[j] = wmma16(a, b, acc[j]);
    }
  }
  int g = ln >> 4, nl = ln & 15;
#pragma unroll
  for (int j = 0; j < 4; ++j) {
    int n = nbase + 16 * j + nl;
    float bias = bkv[n];
#pragma unroll
    for (int r = 0; r < 8; ++r) {
      int m = mbase + r + 8 * g;
      float val = acc[j][r] + bias;
      if (n < CC) {
        kh[(size_t)m * CC + n] = (_Float16)val;
      } else {
        int oc = n - CC;
        int h = oc >> 6, d = oc & 63;
        int b = m / LP, l = m % LP;
        vT[(((size_t)(b * HH + h)) * HD + d) * LP + l] = (_Float16)val;
      }
    }
  }
}

// ---------------------------------------------------------------------------
// Kernel 5: flash attention with additive rpe bias.
// One wave per (b,h, 16-query-row tile): 4000 waves = 500 blocks x 8 waves.
// Key tiles of 32; online softmax; P transposed via per-wave LDS tile
// (pitch 40 halves -> bank-conflict-free b128 reads).
// rpe loads and V fragments are issued at the TOP of each iteration so the
// HBM latency of the 128MB rpe stream hides under the S-WMMAs and the
// exp/shuffle chain.
// ---------------------------------------------------------------------------
__global__ __launch_bounds__(256) void attn(const _Float16* __restrict__ qh,
                                            const _Float16* __restrict__ kh,
                                            const _Float16* __restrict__ vT,
                                            const float* __restrict__ rpe,
                                            float* __restrict__ out) {
  __shared__ _Float16 ldsP[8][16][40];
  int wave = threadIdx.x >> 5;
  int ln   = threadIdx.x & 31;
  int wid  = blockIdx.x * 8 + wave;   // 0..3999
  int mtile = wid % 125;
  int bh    = wid / 125;              // 0..31
  int h = bh & 7, b = bh >> 3;
  int mbase = mtile * 16;
  int g = ln >> 4, nl = ln & 15;

  const _Float16* Q  = qh + (size_t)b * LL * CC + h * HD;   // rows=token, ld=CC
  const _Float16* K  = kh + (size_t)b * LP * CC + h * HD;   // rows=key,   ld=CC
  const _Float16* Vt = vT + (size_t)(b * HH + h) * HD * LP; // rows=hd,    ld=LP
  const float*    rp = rpe + (size_t)h * LL * LL;

  v16h qa0 = loadA(Q, CC, mbase, 0);
  v16h qa1 = loadA(Q, CC, mbase, 32);

  v8f oacc[4] = {};
  float mrow[8], lrow[8];
#pragma unroll
  for (int r = 0; r < 8; ++r) { mrow[r] = -3.0e38f; lrow[r] = 0.0f; }

  for (int kt = 0; kt < LP; kt += 32) {
    int n0 = kt + nl;
    int n1 = n0 + 16;

    // ---- issue all independent loads first: rpe bias tile + V fragments ----
    float bias0[8], bias1[8];
#pragma unroll
    for (int r = 0; r < 8; ++r) {
      int mg = mbase + r + 8 * g;
      bias0[r] = (n0 < LL) ? rp[(size_t)mg * LL + n0] : 0.0f;
      bias1[r] = (n1 < LL) ? rp[(size_t)mg * LL + n1] : 0.0f;
    }
    v16h vb0 = loadB(Vt, LP, 0,  kt);
    v16h vb1 = loadB(Vt, LP, 16, kt);
    v16h vb2 = loadB(Vt, LP, 32, kt);
    v16h vb3 = loadB(Vt, LP, 48, kt);

    // prefetch next rpe tile rows into GL2 (global_prefetch_b8)
    {
      int pn = (kt + 32 < LL) ? kt + 32 + nl : nl;
      __builtin_prefetch(rp + (size_t)(mbase + (ln & 15)) * LL + pn, 0, 0);
    }

    // ---- S = Q K^T for 16 queries x 32 keys (two 16x16 tiles) ----
    v8f s0 = {}, s1 = {};
    s0 = wmma16(qa0, loadB(K, CC, kt,      0),  s0);
    s0 = wmma16(qa1, loadB(K, CC, kt,      32), s0);
    s1 = wmma16(qa0, loadB(K, CC, kt + 16, 0),  s1);
    s1 = wmma16(qa1, loadB(K, CC, kt + 16, 32), s1);

    // ---- online softmax (row reductions within 16-lane groups) ----
#pragma unroll
    for (int r = 0; r < 8; ++r) {
      float v0 = (n0 < LL) ? s0[r] * SCALE + bias0[r] : -3.0e38f;
      float v1 = (n1 < LL) ? s1[r] * SCALE + bias1[r] : -3.0e38f;
      float mx = fmaxf(v0, v1);
#pragma unroll
      for (int o = 8; o; o >>= 1) mx = fmaxf(mx, __shfl_xor(mx, o, 32));
      float mnew  = fmaxf(mrow[r], mx);
      float alpha = __expf(mrow[r] - mnew);
      float p0 = __expf(v0 - mnew);
      float p1 = __expf(v1 - mnew);
      float rs = p0 + p1;
#pragma unroll
      for (int o = 8; o; o >>= 1) rs += __shfl_xor(rs, o, 32);
      lrow[r] = lrow[r] * alpha + rs;
      mrow[r] = mnew;
#pragma unroll
      for (int j = 0; j < 4; ++j) oacc[j][r] *= alpha;
      // stage P (C-layout -> row-major f16) in per-wave LDS tile
      ldsP[wave][r + 8 * g][nl]      = (_Float16)p0;
      ldsP[wave][r + 8 * g][16 + nl] = (_Float16)p1;
    }

    // in-wave LDS ordering: writes above feed the A-fragment reads below
    asm volatile("s_wait_dscnt 0" ::: "memory");

    // load P as a 16x32 A fragment from LDS (two ds_load_b128)
    v16h pa;
    {
      const _Float16* p = &ldsP[wave][ln & 15][(ln >> 4) << 3];
      U16h u;
      u.q[0] = *reinterpret_cast<const uint4*>(p);
      u.q[1] = *reinterpret_cast<const uint4*>(p + 16);
      pa = u.v;
    }

    // ---- O += P @ V (V fragments already in registers) ----
    oacc[0] = wmma16(pa, vb0, oacc[0]);
    oacc[1] = wmma16(pa, vb1, oacc[1]);
    oacc[2] = wmma16(pa, vb2, oacc[2]);
    oacc[3] = wmma16(pa, vb3, oacc[3]);
  }

  // epilogue: out[b][h][d][l] = O / l   (reference's [B,H,hd,L] flattening)
#pragma unroll
  for (int j = 0; j < 4; ++j) {
#pragma unroll
    for (int r = 0; r < 8; ++r) {
      int d = 16 * j + nl;
      int l = mbase + r + 8 * g;
      out[(((size_t)(b * HH + h)) * HD + d) * LL + l] = oacc[j][r] / lrow[r];
    }
  }
}

// ---------------------------------------------------------------------------
// Host launcher
// ---------------------------------------------------------------------------
extern "C" void kernel_launch(void* const* d_in, const int* in_sizes, int n_in,
                              void* d_out, int out_size, void* d_ws, size_t ws_size,
                              hipStream_t stream) {
  const float* x   = (const float*)d_in[0];
  const float* rpe = (const float*)d_in[1];
  const float* Wq  = (const float*)d_in[2];
  const float* bq  = (const float*)d_in[3];
  const float* Wkv = (const float*)d_in[4];
  const float* bkv = (const float*)d_in[5];
  const float* Wl  = (const float*)d_in[6];
  const float* bl  = (const float*)d_in[7];
  float* out = (float*)d_out;

  char* ws = (char*)d_ws;
  size_t off = 0;
  _Float16* xh   = (_Float16*)(ws + off); off += (size_t)BB * LL * CC * 2;
  _Float16* kvin = (_Float16*)(ws + off); off += (size_t)BB * LP * CC * 2;
  _Float16* qh   = (_Float16*)(ws + off); off += (size_t)BB * LL * CC * 2;
  _Float16* kh   = (_Float16*)(ws + off); off += (size_t)BB * LP * CC * 2;
  _Float16* vT   = (_Float16*)(ws + off); off += (size_t)BB * HH * HD * LP * 2;
  _Float16* Wqh  = (_Float16*)(ws + off); off += (size_t)CC * CC * 2;
  _Float16* Wkvh = (_Float16*)(ws + off); off += (size_t)C2 * CC * 2;

  {
    int total = BB * LP * CC;
    prep_conv<<<(total + 255) / 256, 256, 0, stream>>>(x, Wl, bl, xh, kvin);
  }
  {
    int total = CC * CC + C2 * CC;
    prep_weights<<<(total + 255) / 256, 256, 0, stream>>>(Wq, Wkv, Wqh, Wkvh);
  }
  gemm_q<<<500, 256, 0, stream>>>(xh, Wqh, bq, qh);           // 4000 waves
  gemm_kv<<<1008, 256, 0, stream>>>(kvin, Wkvh, bkv, kh, vT); // 8064 waves
  attn<<<500, 256, 0, stream>>>(qh, kh, vT, rpe, out);        // 4000 waves
}
